// XFormerAttention_73624329388177
// MI455X (gfx1250) — matrix operand order repaired and checked by reading, not compiled, not run on hardware
//
#include <hip/hip_runtime.h>
#include <math.h>

// ---------------------------------------------------------------------------
// ProbSparse (Informer-style) attention for MI455X / gfx1250.
// Shapes fixed by the reference: B=2, H=16, T=S=2048, D=64, fp32 in/out.
// u (top-k query count) = max(int(0.4*ln(2048)), int(0.3*2048)) = 614.
//
// Pipeline:
//   A) full QK^T scan (bf16 WMMA, f32 acc) with online softmax stats
//      -> kl score, row max m, row sum l.  K tiles are DMA'd into LDS by the
//      Tensor Data Mover, double-buffered with s_wait_tensorcnt.
//   B) exact top-614 per (b,h) via in-LDS bitonic sort.
//   C) sparse attention for selected rows, reusing (m,l) from A (identical
//      scores) -> single pass, no flash rescale.  K/V tiles staged by TDM.
// ---------------------------------------------------------------------------

#define B_      2
#define H_      16
#define T_      2048
#define S_      2048
#define D_      64
#define BH_     (B_ * H_)
#define U_      614
#define NT_     ((U_ + 15) / 16)          // 39 query tiles in sparse pass
#define SCALE_  0.125f                    // 1/sqrt(64)
#define LOG_S_  7.6246189861593985f       // ln(2048)
#define KSTEP_  32                        // keys per pipeline step
#define NSTEP_  (S_ / KSTEP_)             // 64

typedef __bf16 bf16_t;
typedef __attribute__((ext_vector_type(16))) __bf16 v16bf;
typedef __attribute__((ext_vector_type(8)))  float  v8f;

// ---------------------------------------------------------------------------
// Tensor Data Mover: DMA a [rows x 64] f32 tile (row stride 64) into LDS.
// Descriptor per CDNA5 ISA ch.8 (group0: count/lds/global/type=2,
// group1: data_size=4B, tensor_dim0=64, tile_dim0=64, tile_dim1=rows,
// tensor_dim0_stride=64).  Tracked by TENSORcnt.
// ---------------------------------------------------------------------------
#if defined(__HIP_DEVICE_COMPILE__) && __has_builtin(__builtin_amdgcn_tensor_load_to_lds)
#define USE_TDM 1
#else
#define USE_TDM 0
#endif

#if USE_TDM
typedef unsigned int u32_t;
typedef __attribute__((ext_vector_type(4))) unsigned int v4u;
typedef __attribute__((ext_vector_type(4))) int v4i;
typedef __attribute__((ext_vector_type(8))) int v8i;

__device__ __forceinline__ unsigned lds_offset(const void* p) {
  // LDS aperture: flat addr [31:0] is the wave-relative LDS byte address.
  return (unsigned)(unsigned long long)p;
}

__device__ __forceinline__ void tdm_load_tile_f32(unsigned lds_byte_off,
                                                  const float* __restrict__ gsrc,
                                                  int rows, int tensor_rows) {
  const unsigned long long ga = (unsigned long long)gsrc;
  v4u g0;
  g0[0] = 1u;                                            // count=1, user mode
  g0[1] = lds_byte_off;                                  // lds_addr (bytes)
  g0[2] = (u32_t)ga;                                     // global_addr[31:0]
  g0[3] = (u32_t)((ga >> 32) & 0x01FFFFFFu) | (2u << 30);// addr[56:32] | type=2
  v8i g1;
  g1[0] = (int)(2u << 16);                               // data_size = 4 bytes
  g1[1] = (int)((u32_t)D_ << 16);                        // tensor_dim0 = 64 (lo16)
  g1[2] = (int)(((u32_t)tensor_rows & 0xFFFFu) << 16);   // tensor_dim1 lo16
  g1[3] = (int)(((u32_t)D_ << 16) |
                (((u32_t)tensor_rows >> 16) & 0xFFFFu)); // dim1 hi16 | tile_dim0=64
  g1[4] = rows & 0xFFFF;                                 // tile_dim1=rows, tile_dim2=0
  g1[5] = D_;                                            // tensor_dim0_stride = 64
  g1[6] = 0;
  g1[7] = 0;
  const v4i z4 = {0, 0, 0, 0};
#if __clang_major__ >= 23
  const v8i z8 = {0, 0, 0, 0, 0, 0, 0, 0};
  __builtin_amdgcn_tensor_load_to_lds(g0, g1, z4, z4, z8, 0);
#else
  __builtin_amdgcn_tensor_load_to_lds(g0, g1, z4, z4, 0);
#endif
}
#endif  // USE_TDM

// ---------------------------------------------------------------------------
// WMMA helpers.  D = A(16x32 bf16) * B(32x16 bf16) + C(16x16 f32)
// ---------------------------------------------------------------------------
__device__ __forceinline__ v8f wmma_bf16_acc(v16bf a, v16bf b, v8f c) {
  return __builtin_amdgcn_wmma_f32_16x16x32_bf16(false, a, false, b,
                                                 (short)0, c, false, false);
}

// B-fragment (32x16): lane holds N = lane&15; element j -> K = j + 16*(lane>>4).
// Source: 16 consecutive floats at the lane's (row, col-base).
__device__ __forceinline__ v16bf load_bfrag16(const float* __restrict__ p) {
  v16bf r;
#pragma unroll
  for (int j = 0; j < 16; ++j) r[j] = (bf16_t)p[j];
  return r;
}

// A-fragment (16x32): lane holds M = lane&15; elements 0..7 -> K = j + 16*half,
// elements 8..15 -> K = 8 + (j-8) + 16*half.  Two 8-float contiguous blocks.
__device__ __forceinline__ v16bf load_afrag16(const float* __restrict__ plo,
                                              const float* __restrict__ phi) {
  v16bf r;
#pragma unroll
  for (int j = 0; j < 8; ++j) r[j] = (bf16_t)plo[j];
#pragma unroll
  for (int j = 0; j < 8; ++j) r[8 + j] = (bf16_t)phi[j];
  return r;
}

// ---------------------------------------------------------------------------
// Kernel A: full QK^T scan with online softmax stats.
// One wave per (bh, 16-query tile); 32 keys per step, TDM double buffered.
// C-fragment layout: VGPR i, lane (h=lane>>4, l=lane&15): row = i+8h, col = l.
// Per-lane partial (m,l,t) over its column subset, xor-shuffle merged at end.
// ---------------------------------------------------------------------------
__global__ __launch_bounds__(32) void kl_stats_kernel(
    const float* __restrict__ q, const float* __restrict__ k,
    float* __restrict__ ws_kl, float* __restrict__ ws_m,
    float* __restrict__ ws_l) {
  __shared__ float lds_kt[2][KSTEP_ * D_];   // 2 x 8 KB double buffer

  const int bh = blockIdx.x >> 7;            // 128 query tiles per (b,h)
  const int q0 = (blockIdx.x & 127) << 4;
  const int ln = threadIdx.x;
  const int lh = ln >> 4;
  const int ll = ln & 15;

  const float* qb = q + ((size_t)bh * T_ + q0) * D_;
  const float* kb = k + (size_t)bh * S_ * D_;

  const float* qrow = qb + (size_t)ll * D_;
  const v16bf aq0 = load_afrag16(qrow + 16 * lh,      qrow + 8 + 16 * lh);
  const v16bf aq1 = load_afrag16(qrow + 32 + 16 * lh, qrow + 40 + 16 * lh);

  float mr[8], lr[8], tr[8];
#pragma unroll
  for (int i = 0; i < 8; ++i) { mr[i] = -1e30f; lr[i] = 0.f; tr[i] = 0.f; }

  const v8f vzero = {0.f, 0.f, 0.f, 0.f, 0.f, 0.f, 0.f, 0.f};

#if USE_TDM
  tdm_load_tile_f32(lds_offset(&lds_kt[0][0]), kb, KSTEP_, S_);
#endif

  for (int step = 0; step < NSTEP_; ++step) {
    const float* cbuf = &lds_kt[step & 1][0];

#if USE_TDM
    if (step + 1 < NSTEP_) {
      tdm_load_tile_f32(lds_offset(&lds_kt[(step + 1) & 1][0]),
                        kb + (size_t)(step + 1) * KSTEP_ * D_, KSTEP_, S_);
      __builtin_amdgcn_s_wait_tensorcnt(1);   // oldest (current buffer) done
    } else {
      __builtin_amdgcn_s_wait_tensorcnt(0);
    }
#else
    {  // fallback: cooperative coalesced staging
      float* dst = &lds_kt[step & 1][0];
      for (int it = ln; it < KSTEP_ * (D_ / 4); it += 32) {
        const int row = it >> 4, c4 = it & 15;
        ((float4*)dst)[it] =
            ((const float4*)(kb + (size_t)(step * KSTEP_ + row) * D_))[c4];
      }
    }
#endif
    __syncthreads();

    // two 16-key subtiles -> two C fragments
    v8f acc[2];
#pragma unroll
    for (int kt = 0; kt < 2; ++kt) {
      const float* krow = cbuf + (16 * kt + ll) * D_ + 16 * lh;
      const v16bf bk0 = load_bfrag16(krow);
      const v16bf bk1 = load_bfrag16(krow + 32);
      acc[kt] = vzero;
      acc[kt] = wmma_bf16_acc(aq0, bk0, acc[kt]);
      acc[kt] = wmma_bf16_acc(aq1, bk1, acc[kt]);
    }

    // merged online update: one rescale exp per two score elements
#pragma unroll
    for (int i = 0; i < 8; ++i) {
      const float sa = fminf(fmaxf(acc[0][i] * SCALE_, -50.f), 50.f);
      const float sb = fminf(fmaxf(acc[1][i] * SCALE_, -50.f), 50.f);
      const float mo = mr[i];
      const float mn = fmaxf(mo, fmaxf(sa, sb));
      const float cr = __expf(mo - mn);
      const float ea = __expf(sa - mn);
      const float eb = __expf(sb - mn);
      const float lo = lr[i];
      tr[i] = cr * (tr[i] + (mo - mn) * lo) + ea * (sa - mn) + eb * (sb - mn);
      lr[i] = cr * lo + ea + eb;
      mr[i] = mn;
    }
    __syncthreads();
  }

  // merge the 16 per-lane partials inside each half-wave
#pragma unroll
  for (int mask = 1; mask <= 8; mask <<= 1) {
#pragma unroll
    for (int i = 0; i < 8; ++i) {
      float om = __shfl_xor(mr[i], mask, 32);
      float ol = __shfl_xor(lr[i], mask, 32);
      float ot = __shfl_xor(tr[i], mask, 32);
      float mn = fmaxf(mr[i], om);
      float c1 = __expf(mr[i] - mn);
      float c2 = __expf(om - mn);
      tr[i] = c1 * (tr[i] + (mr[i] - mn) * lr[i]) + c2 * (ot + (om - mn) * ol);
      lr[i] = c1 * lr[i] + c2 * ol;
      mr[i] = mn;
    }
  }

  if (ll == 0) {                       // lane 0 -> rows 0..7, lane 16 -> 8..15
#pragma unroll
    for (int i = 0; i < 8; ++i) {
      const int row = q0 + i + 8 * lh;
      const size_t o = (size_t)bh * T_ + row;
      const float ls = lr[i];
      ws_kl[o] = tr[i] / ls - __logf(ls) + LOG_S_;   // sum p log p + log S
      ws_m[o]  = mr[i];
      ws_l[o]  = ls;
    }
  }
}

// ---------------------------------------------------------------------------
// Kernel B: exact top-614 per (b,h) via in-LDS bitonic sort of 2048 packed
// keys (monotone float bits ‖ ~index) sorted descending; tie-break = lower idx.
// ---------------------------------------------------------------------------
__global__ __launch_bounds__(256) void topk_kernel(
    const float* __restrict__ ws_kl, int* __restrict__ ws_topk) {
  __shared__ unsigned long long keys[T_];
  const int bh  = blockIdx.x;
  const int tid = threadIdx.x;

  for (int i = tid; i < T_; i += 256) {
    unsigned int ub = __float_as_uint(ws_kl[(size_t)bh * T_ + i]);
    ub = (ub & 0x80000000u) ? ~ub : (ub | 0x80000000u);   // order-preserving map
    keys[i] = ((unsigned long long)ub << 32) |
              (unsigned long long)(0xFFFFFFFFu - (unsigned int)i);
  }
  __syncthreads();

  for (int kk = 2; kk <= T_; kk <<= 1) {
    for (int j = kk >> 1; j > 0; j >>= 1) {
      for (int i = tid; i < T_; i += 256) {
        const int p = i ^ j;
        if (p > i) {
          const unsigned long long a = keys[i];
          const unsigned long long b = keys[p];
          const bool sw = ((i & kk) == 0) ? (a < b) : (a > b);  // descending
          if (sw) { keys[i] = b; keys[p] = a; }
        }
      }
      __syncthreads();
    }
  }

  for (int i = tid; i < U_; i += 256)
    ws_topk[bh * U_ + i] =
        (int)(0xFFFFFFFFu - (unsigned int)(keys[i] & 0xFFFFFFFFull));
}

// ---------------------------------------------------------------------------
// Zero-fill output (harness poisons d_out with 0xAA).
// ---------------------------------------------------------------------------
__global__ void fill_zero_kernel(float4* __restrict__ out, int n4) {
  const int i = blockIdx.x * blockDim.x + threadIdx.x;
  if (i < n4) out[i] = make_float4(0.f, 0.f, 0.f, 0.f);
}

// ---------------------------------------------------------------------------
// Kernel C: sparse attention for selected queries.  One wave per 16 selected
// rows; K/V tiles DMA'd into LDS by the TDM, probabilities transposed through
// LDS into A-layout, P.V done with K=32 bf16 WMMAs accumulating f32 output.
// Uses m,l from kernel A (identical scores) -> single pass, no rescale.
// ---------------------------------------------------------------------------
__global__ __launch_bounds__(32) void sparse_attn_kernel(
    const float* __restrict__ q, const float* __restrict__ k,
    const float* __restrict__ v, const float* __restrict__ ws_m,
    const float* __restrict__ ws_l, const int* __restrict__ ws_topk,
    float* __restrict__ out) {
  __shared__ float lds_k[KSTEP_ * D_];   // 8 KB
  __shared__ float lds_v[KSTEP_ * D_];   // 8 KB
  __shared__ float lds_p[16 * KSTEP_];   // 2 KB

  const int bh   = blockIdx.x / NT_;
  const int tile = blockIdx.x % NT_;
  const int base = tile * 16;
  const int ln = threadIdx.x;
  const int lh = ln >> 4;
  const int ll = ln & 15;

  const float* qb = q + (size_t)bh * T_ * D_;
  const float* kb = k + (size_t)bh * S_ * D_;
  const float* vb = v + (size_t)bh * S_ * D_;
  const int*   tk = ws_topk + bh * U_;

  // gather Q rows in A-layout (M = lane&15)
  const int rA  = base + ll;
  const int qiA = (rA < U_) ? tk[rA] : 0;
  const float* qrow = qb + (size_t)qiA * D_;
  const v16bf aq0 = load_afrag16(qrow + 16 * lh,      qrow + 8 + 16 * lh);
  const v16bf aq1 = load_afrag16(qrow + 32 + 16 * lh, qrow + 40 + 16 * lh);

  // softmax stats + scatter indices for the C-layout rows (row = i + 8h)
  float mi[8], li[8];
  int   qiC[8];
  bool  ok[8];
#pragma unroll
  for (int i = 0; i < 8; ++i) {
    const int r = base + i + 8 * lh;
    ok[i]  = (r < U_);
    const int qi = ok[i] ? tk[r] : 0;
    qiC[i] = qi;
    mi[i] = ws_m[(size_t)bh * T_ + qi];
    li[i] = ws_l[(size_t)bh * T_ + qi];
  }

  const v8f vzero = {0.f, 0.f, 0.f, 0.f, 0.f, 0.f, 0.f, 0.f};
  v8f oacc[4];
#pragma unroll
  for (int c = 0; c < 4; ++c) oacc[c] = vzero;

  for (int s0 = 0; s0 < S_; s0 += KSTEP_) {
#if USE_TDM
    tdm_load_tile_f32(lds_offset(lds_k), kb + (size_t)s0 * D_, KSTEP_, S_);
    tdm_load_tile_f32(lds_offset(lds_v), vb + (size_t)s0 * D_, KSTEP_, S_);
    __builtin_amdgcn_s_wait_tensorcnt(0);
#else
    for (int it = ln; it < KSTEP_ * (D_ / 4); it += 32) {
      const int row = it >> 4, c4 = it & 15;
      ((float4*)lds_k)[it] = ((const float4*)(kb + (size_t)(s0 + row) * D_))[c4];
      ((float4*)lds_v)[it] = ((const float4*)(vb + (size_t)(s0 + row) * D_))[c4];
    }
#endif
    __syncthreads();

    // scores for two 16-key subtiles -> clipped probabilities into lds_p
#pragma unroll
    for (int kt = 0; kt < 2; ++kt) {
      const float* krow = lds_k + (16 * kt + ll) * D_ + 16 * lh;
      const v16bf bk0 = load_bfrag16(krow);
      const v16bf bk1 = load_bfrag16(krow + 32);
      v8f acc = vzero;
      acc = wmma_bf16_acc(aq0, bk0, acc);
      acc = wmma_bf16_acc(aq1, bk1, acc);
#pragma unroll
      for (int i = 0; i < 8; ++i) {
        const float s = fminf(fmaxf(acc[i] * SCALE_, -50.f), 50.f);
        float w = __expf(s - mi[i]) / li[i];
        w = fminf(fmaxf(w, 1e-9f), 1.0f);
        lds_p[(i + 8 * lh) * KSTEP_ + 16 * kt + ll] = w;
      }
    }
    __syncthreads();

    // P (16x32, A-layout) x V (32x64, 4 B-frags of 16 cols) += O
    const float* prow = lds_p + ll * KSTEP_;
    const v16bf ap = load_afrag16(prow + 16 * lh, prow + 8 + 16 * lh);
#pragma unroll
    for (int c = 0; c < 4; ++c) {
      v16bf bv;
#pragma unroll
      for (int j = 0; j < 16; ++j)
        bv[j] = (bf16_t)lds_v[(j + 16 * lh) * D_ + 16 * c + ll];
      oacc[c] = wmma_bf16_acc(ap, bv, oacc[c]);
    }
    __syncthreads();   // protect LDS before next stage
  }

  // scatter O rows to the selected query positions
  float* ob = out + (size_t)bh * T_ * D_;
#pragma unroll
  for (int i = 0; i < 8; ++i) {
    if (ok[i]) {
      float* orow = ob + (size_t)qiC[i] * D_ + ll;
#pragma unroll
      for (int c = 0; c < 4; ++c) orow[16 * c] = oacc[c][i];
    }
  }
}

// ---------------------------------------------------------------------------
// Host entry point.
// Workspace layout (floats): kl[32*2048] | m[32*2048] | l[32*2048] | topk int[32*614]
// ~865 KB total.
// ---------------------------------------------------------------------------
extern "C" void kernel_launch(void* const* d_in, const int* in_sizes, int n_in,
                              void* d_out, int out_size, void* d_ws,
                              size_t ws_size, hipStream_t stream) {
  (void)in_sizes; (void)n_in; (void)out_size; (void)ws_size;
  const float* q = (const float*)d_in[0];
  const float* k = (const float*)d_in[1];
  const float* v = (const float*)d_in[2];
  float* out = (float*)d_out;

  float* ws    = (float*)d_ws;
  float* ws_kl = ws;
  float* ws_m  = ws + (size_t)BH_ * T_;
  float* ws_l  = ws + 2 * (size_t)BH_ * T_;
  int*   ws_tk = (int*)(ws + 3 * (size_t)BH_ * T_);

  const int n4 = (B_ * H_ * T_ * D_) / 4;
  fill_zero_kernel<<<(n4 + 255) / 256, 256, 0, stream>>>((float4*)out, n4);
  kl_stats_kernel<<<BH_ * (T_ / 16), 32, 0, stream>>>(q, k, ws_kl, ws_m, ws_l);
  topk_kernel<<<BH_, 256, 0, stream>>>(ws_kl, ws_tk);
  sparse_attn_kernel<<<BH_ * NT_, 32, 0, stream>>>(q, k, v, ws_m, ws_l, ws_tk,
                                                   out);
}
